// MPNNNet_22754736734327
// MI455X (gfx1250) — compile-verified, matrix operand order checked by reading, not run on hardware
//
#include <hip/hip_runtime.h>
#include <hip/hip_bf16.h>

typedef __attribute__((ext_vector_type(16))) _Float16 v16h;
typedef __attribute__((ext_vector_type(8)))  _Float16 v8h;
typedef __attribute__((ext_vector_type(8)))  float    v8f;

#define NN   5000
#define EE   20000
#define GG   128
#define HH   64
#define LL   3
// K-chunks per edge-tile contraction: 128 k-values x 2 h-halves + 2 bias chunks
#define NCHUNK 258

// ---------------------------------------------------------------------------
// gfx1250 async global->LDS copy (ASYNCcnt-tracked). VDST = 32-bit LDS byte
// address (low 32 bits of the generic pointer, per ISA 10.2 LDS aperture).
__device__ __forceinline__ void async_g2l_b128(void* lds_ptr, const void* gptr) {
  unsigned l = (unsigned)(unsigned long long)lds_ptr;
  asm volatile("global_load_async_to_lds_b128 %0, %1, off"
               :: "v"(l), "v"(gptr) : "memory");
}

// ---------------------------------------------------------------------------
// zero helper
__global__ void zero_f32(float* __restrict__ p, long n) {
  long i = (long)blockIdx.x * blockDim.x + threadIdx.x;
  if (i < n) p[i] = 0.0f;
}

// ---------------------------------------------------------------------------
// input projection: h = x @ in_W + in_b ; also initializes GRU hidden state
__global__ void in_proj(const float* __restrict__ x, const float* __restrict__ W,
                        const float* __restrict__ b, float* __restrict__ hcur,
                        float* __restrict__ hg) {
  int idx = blockIdx.x * blockDim.x + threadIdx.x;
  if (idx >= NN * HH) return;
  int n = idx >> 6, j = idx & 63;
  const float* row = x + (long)n * 32;
  float acc = b[j];
#pragma unroll
  for (int i = 0; i < 32; ++i) acc += row[i] * W[i * HH + j];
  hcur[idx] = acc;
  hg[idx] = acc;
}

// ---------------------------------------------------------------------------
// f32 -> f16 node features (for WMMA A-side gather)
__global__ void h_to_f16(const float* __restrict__ h, _Float16* __restrict__ h16) {
  int idx = blockIdx.x * blockDim.x + threadIdx.x;
  if (idx < NN * HH) h16[idx] = (_Float16)h[idx];
}

// ---------------------------------------------------------------------------
// edge MLP stage 1: eAct = relu(edge_attr @ em_W1[l] + em_b1[l]) stored as f16
__global__ __launch_bounds__(256) void edge_mlp(const float* __restrict__ ea,
                                                const float* __restrict__ W1,
                                                const float* __restrict__ b1,
                                                _Float16* __restrict__ eact) {
  long idx = (long)blockIdx.x * 256 + threadIdx.x;
  if (idx >= (long)EE * 128) return;
  long e = idx >> 7;
  int o = (int)(idx & 127);
  const float* row = ea + e * 16;
  float acc = b1[o];
#pragma unroll
  for (int i = 0; i < 16; ++i) acc += row[i] * W1[i * 128 + o];
  eact[idx] = (_Float16)fmaxf(acc, 0.0f);
}

// ---------------------------------------------------------------------------
// Pre-swizzle em_W2 (+ em_b2 as 2 extra K-chunks) into WMMA-B register order:
// tile t = ((l*258 + c)*4 + oc), 512 halves; position = lane*16 + r,
// K = (r/8)*16 + (lane/16)*8 + (r%8), N = lane%16  (mirrors 16-bit A layout)
__global__ void swizzle_w2(const float* __restrict__ W2, const float* __restrict__ b2,
                           _Float16* __restrict__ out) {
  long idx = (long)blockIdx.x * blockDim.x + threadIdx.x;
  const long total = (long)LL * NCHUNK * 4 * 512;
  if (idx >= total) return;
  long t = idx >> 9;
  int pos = (int)(idx & 511);
  int lane = pos >> 4, r = pos & 15;
  int l = (int)(t / (NCHUNK * 4));
  int rem = (int)(t % (NCHUNK * 4));
  int c = rem >> 2, oc = rem & 3;
  int K = (r >> 3) * 16 + (lane >> 4) * 8 + (r & 7);
  int o = oc * 16 + (lane & 15);
  float val;
  if (c < 256) {
    int k = c >> 1, hc = c & 1;
    int h = hc * 32 + K;
    val = W2[(long)l * 128 * 4096 + (long)k * 4096 + h * 64 + o];
  } else {
    int h = (c & 1) * 32 + K;
    val = b2[(long)l * 4096 + h * 64 + o];
  }
  out[idx] = (_Float16)val;
}

// ---------------------------------------------------------------------------
// Fused NNConv message + scatter-add.
// One wave (32 threads) per 16-edge tile; the wave owns all 4 output chunks
// (4 f32 accumulators). msg[16,64] = Z[16,8256] @ W2flat[8256,64] via
// 258 iterations x 4 v_wmma_f32_16x16x32_f16. The B stream is double-buffered
// in LDS with gfx1250 async global->LDS copies, throttled by s_wait_asynccnt.
__global__ __launch_bounds__(32) void nnconv_wmma(
    const _Float16* __restrict__ h16g, const _Float16* __restrict__ eact16,
    const _Float16* __restrict__ w2h,  // this layer's swizzled slab
    const int* __restrict__ edge_index, float* __restrict__ agg) {
  __shared__ __align__(32) _Float16 hT[16 * 64];     // gathered h_src, f16
  __shared__ __align__(32) _Float16 eT[16 * 128];    // eAct, f16
  __shared__ __align__(32) _Float16 Bb[2][4 * 512];  // double-buffered B chunks
  __shared__ int dT[16];

  const int lane = threadIdx.x;
  const long eBase = (long)blockIdx.x * 16;

  // stage gathered h_src (128 x v8h) and eAct (128 x v16h); single wave -> LDS
  // ops stay in order, no barrier needed.
#pragma unroll
  for (int i = 0; i < 4; ++i) {
    int chunk = lane + i * 32;  // 0..127
    int e = chunk >> 3, p = chunk & 7;
    int src = edge_index[eBase + e];
    *(v8h*)&hT[e * 64 + p * 8] = *(const v8h*)&h16g[(long)src * 64 + p * 8];
  }
#pragma unroll
  for (int i = 0; i < 4; ++i) {
    int chunk = lane + i * 32;
    *(v16h*)&eT[chunk * 16] = *(const v16h*)&eact16[eBase * 128 + chunk * 16];
  }
  if (lane < 16) dT[lane] = edge_index[EE + eBase + lane];
  __syncthreads();

  const int n16 = lane & 15;     // row-of-16 within A/C layouts
  const int hiHalf = lane >> 4;  // K sub-bank select

  // prologue: async-load chunk 0 (4KB = 8 x b128 across the wave)
#pragma unroll
  for (int i = 0; i < 8; ++i)
    async_g2l_b128(&Bb[0][i * 256 + lane * 8], &w2h[i * 256 + lane * 8]);

  v8f acc0 = {}, acc1 = {}, acc2 = {}, acc3 = {};
  for (int c = 0; c < NCHUNK; ++c) {
    if (c + 1 < NCHUNK) {
      const _Float16* gsrc = &w2h[(long)(c + 1) * 2048];
      _Float16* ldst = &Bb[(c + 1) & 1][0];
#pragma unroll
      for (int i = 0; i < 8; ++i)
        async_g2l_b128(&ldst[i * 256 + lane * 8], &gsrc[i * 256 + lane * 8]);
      // newest batch (8 ops) may fly; the batch for chunk c must have landed
      asm volatile("s_wait_asynccnt 0x8" ::: "memory");
    } else {
      asm volatile("s_wait_asynccnt 0x0" ::: "memory");
    }

    // A tile: Z[e, j] = eAct[e, c/2] * h_src[e, (c&1)*32 + K(lane,j)]
    const int hb = (c & 1) * 32 + hiHalf * 8;
    v8h lo = *(const v8h*)&hT[n16 * 64 + hb];
    v8h hi = *(const v8h*)&hT[n16 * 64 + hb + 16];
    v16h a = __builtin_shufflevector(lo, hi, 0, 1, 2, 3, 4, 5, 6, 7, 8, 9, 10,
                                     11, 12, 13, 14, 15);
    _Float16 s = (c < 256) ? eT[n16 * 128 + (c >> 1)] : (_Float16)1.0f;
    v16h sv;
#pragma unroll
    for (int q = 0; q < 16; ++q) sv[q] = s;
    a = a * sv;  // packed f16 multiplies, co-execute with WMMA

    const _Float16* bb = &Bb[c & 1][0];
    v16h b0 = *(const v16h*)&bb[0 * 512 + lane * 16];
    v16h b1 = *(const v16h*)&bb[1 * 512 + lane * 16];
    v16h b2 = *(const v16h*)&bb[2 * 512 + lane * 16];
    v16h b3 = *(const v16h*)&bb[3 * 512 + lane * 16];
    acc0 = __builtin_amdgcn_wmma_f32_16x16x32_f16(false, a, false, b0, (short)0,
                                                  acc0, false, false);
    acc1 = __builtin_amdgcn_wmma_f32_16x16x32_f16(false, a, false, b1, (short)0,
                                                  acc1, false, false);
    acc2 = __builtin_amdgcn_wmma_f32_16x16x32_f16(false, a, false, b2, (short)0,
                                                  acc2, false, false);
    acc3 = __builtin_amdgcn_wmma_f32_16x16x32_f16(false, a, false, b3, (short)0,
                                                  acc3, false, false);
  }

  // scatter-add msg into agg[dst] (C layout: lane<16 -> M=v, lane>=16 -> M=v+8)
#pragma unroll
  for (int v = 0; v < 8; ++v) {
    int m = hiHalf * 8 + v;
    long base = (long)dT[m] * HH;
    atomicAdd(&agg[base + 0 * 16 + n16], acc0[v]);
    atomicAdd(&agg[base + 1 * 16 + n16], acc1[v]);
    atomicAdd(&agg[base + 2 * 16 + n16], acc2[v]);
    atomicAdd(&agg[base + 3 * 16 + n16], acc3[v]);
  }
}

// ---------------------------------------------------------------------------
// m = agg + h @ root_W[l] + conv_b[l]   (in-place on agg)
__global__ __launch_bounds__(256) void conv_root(const float* __restrict__ h,
                                                 float* __restrict__ agg,
                                                 const float* __restrict__ rootW,
                                                 const float* __restrict__ convb) {
  __shared__ float W[64 * 64];
  __shared__ float hrow[4][64];
  int tid = threadIdx.x;
  for (int i = tid; i < 4096; i += 256) W[i] = rootW[i];
  int nb = blockIdx.x * 4;
  int ln = tid >> 6, o = tid & 63;
  if (nb + ln < NN) hrow[ln][o] = h[(long)(nb + ln) * HH + o];
  __syncthreads();
  int n = nb + ln;
  if (n >= NN) return;
  float acc = convb[o];
#pragma unroll 8
  for (int k = 0; k < 64; ++k) acc += hrow[ln][k] * W[k * 64 + o];
  agg[(long)n * HH + o] += acc;
}

// ---------------------------------------------------------------------------
// one GRU step (PyTorch gate order r,z,n) + BN statistics accumulation
__global__ __launch_bounds__(64) void gru_step(const float* __restrict__ m,
                                               float* __restrict__ hg,
                                               const float* __restrict__ Wih,
                                               const float* __restrict__ Whh,
                                               const float* __restrict__ bih,
                                               const float* __restrict__ bhh,
                                               float* __restrict__ stats) {
  __shared__ float ms[64], hs[64];
  int n = blockIdx.x, j = threadIdx.x;
  ms[j] = m[(long)n * HH + j];
  hs[j] = hg[(long)n * HH + j];
  __syncthreads();
  float ir = bih[j], iz = bih[64 + j], in_ = bih[128 + j];
  float hr = bhh[j], hz = bhh[64 + j], hn = bhh[128 + j];
#pragma unroll 4
  for (int k = 0; k < 64; ++k) {
    float mk = ms[k], hk = hs[k];
    ir += mk * Wih[j * 64 + k];
    iz += mk * Wih[(64 + j) * 64 + k];
    in_ += mk * Wih[(128 + j) * 64 + k];
    hr += hk * Whh[j * 64 + k];
    hz += hk * Whh[(64 + j) * 64 + k];
    hn += hk * Whh[(128 + j) * 64 + k];
  }
  float r = 1.0f / (1.0f + __expf(-(ir + hr)));
  float z = 1.0f / (1.0f + __expf(-(iz + hz)));
  float nc = tanhf(in_ + r * hn);
  float hnew = (1.0f - z) * nc + z * hs[j];
  hg[(long)n * HH + j] = hnew;
  atomicAdd(&stats[j], hnew);
  atomicAdd(&stats[64 + j], hnew * hnew);
}

// ---------------------------------------------------------------------------
// BatchNorm(train stats) + ReLU + residual (residual source = layer input h)
__global__ void bn_relu_res(const float* __restrict__ hg,
                            const float* __restrict__ stats,
                            const float* __restrict__ g, const float* __restrict__ b,
                            const float* __restrict__ hres, float* __restrict__ hout) {
  int idx = blockIdx.x * blockDim.x + threadIdx.x;
  if (idx >= NN * HH) return;
  int j = idx & 63;
  const float invN = 1.0f / (float)NN;
  float mean = stats[j] * invN;
  float var = stats[64 + j] * invN - mean * mean;
  float v = g[j] * (hg[idx] - mean) * rsqrtf(var + 1e-5f) + b[j];
  hout[idx] = fmaxf(v, 0.0f) + hres[idx];
}

// ---------------------------------------------------------------------------
// global mean pool accumulation
__global__ void pool_accum(const float* __restrict__ h, const int* __restrict__ batch,
                           float* __restrict__ sums, float* __restrict__ cnt) {
  int idx = blockIdx.x * blockDim.x + threadIdx.x;
  if (idx >= NN * HH) return;
  int n = idx >> 6, j = idx & 63;
  int bg = batch[n];
  atomicAdd(&sums[(long)bg * HH + j], h[idx]);
  if (j == 0) atomicAdd(&cnt[bg], 1.0f);
}

// ---------------------------------------------------------------------------
// readout: Linear -> BN(over 128 graphs) -> ReLU -> Linear, one workgroup
__global__ __launch_bounds__(128) void readout(const float* __restrict__ sums,
                                               const float* __restrict__ cnt,
                                               const float* __restrict__ W1,
                                               const float* __restrict__ b1,
                                               const float* __restrict__ bng,
                                               const float* __restrict__ bnb,
                                               const float* __restrict__ W2,
                                               const float* __restrict__ b2,
                                               float* __restrict__ out) {
  __shared__ float r1[GG * HH];
  __shared__ float mean[HH], istd[HH];
  int t = threadIdx.x;  // graph id
  float gv[64];
  float c = fmaxf(cnt[t], 1.0f);
#pragma unroll
  for (int h = 0; h < 64; ++h) gv[h] = sums[(long)t * HH + h] / c;
  for (int j = 0; j < 64; ++j) {
    float acc = b1[j];
#pragma unroll 8
    for (int h = 0; h < 64; ++h) acc += gv[h] * W1[h * 64 + j];
    r1[t * 64 + j] = acc;
  }
  __syncthreads();
  if (t < 64) {
    float s = 0.0f, s2 = 0.0f;
    for (int r = 0; r < GG; ++r) {
      float v = r1[r * 64 + t];
      s += v;
      s2 += v * v;
    }
    float mm = s / (float)GG;
    mean[t] = mm;
    istd[t] = rsqrtf(s2 / (float)GG - mm * mm + 1e-5f);
  }
  __syncthreads();
  float acc = b2[0];
  for (int j = 0; j < 64; ++j) {
    float v = bng[j] * (r1[t * 64 + j] - mean[j]) * istd[j] + bnb[j];
    acc += fmaxf(v, 0.0f) * W2[j];
  }
  out[t] = acc;
}

// ---------------------------------------------------------------------------
extern "C" void kernel_launch(void* const* d_in, const int* in_sizes, int n_in,
                              void* d_out, int out_size, void* d_ws, size_t ws_size,
                              hipStream_t stream) {
  (void)in_sizes; (void)n_in; (void)out_size; (void)ws_size;
  const float* x        = (const float*)d_in[0];
  const float* ea       = (const float*)d_in[1];
  const int*   eidx     = (const int*)d_in[2];
  const int*   batch    = (const int*)d_in[3];
  const float* in_W     = (const float*)d_in[4];
  const float* in_b     = (const float*)d_in[5];
  const float* em_W1    = (const float*)d_in[6];
  const float* em_b1    = (const float*)d_in[7];
  const float* em_W2    = (const float*)d_in[8];
  const float* em_b2    = (const float*)d_in[9];
  const float* root_W   = (const float*)d_in[10];
  const float* conv_b   = (const float*)d_in[11];
  const float* bn_g     = (const float*)d_in[12];
  const float* bn_b     = (const float*)d_in[13];
  const float* gru_Wih  = (const float*)d_in[14];
  const float* gru_Whh  = (const float*)d_in[15];
  const float* gru_bih  = (const float*)d_in[16];
  const float* gru_bhh  = (const float*)d_in[17];
  const float* ro_W1    = (const float*)d_in[18];
  const float* ro_b1    = (const float*)d_in[19];
  const float* ro_bng   = (const float*)d_in[20];
  const float* ro_bnb   = (const float*)d_in[21];
  const float* ro_W2    = (const float*)d_in[22];
  const float* ro_b2    = (const float*)d_in[23];
  float* out = (float*)d_out;

  // ---- carve workspace ----
  char* ws = (char*)d_ws;
  size_t off = 0;
  auto carve = [&](size_t bytes) {
    void* p = ws + off;
    off = (off + bytes + 255) & ~(size_t)255;
    return p;
  };
  float*     h_cur  = (float*)carve((size_t)NN * HH * 4);
  float*     h_nxt  = (float*)carve((size_t)NN * HH * 4);
  float*     hg     = (float*)carve((size_t)NN * HH * 4);
  float*     agg    = (float*)carve((size_t)NN * HH * 4);
  _Float16*  h16    = (_Float16*)carve((size_t)NN * HH * 2);
  _Float16*  eact16 = (_Float16*)carve((size_t)EE * 128 * 2);
  _Float16*  w2h    = (_Float16*)carve((size_t)LL * NCHUNK * 4 * 512 * 2);
  float*     stats  = (float*)carve(128 * 4);
  float*     pool   = (float*)carve((size_t)GG * HH * 4);
  float*     cnt    = (float*)carve(GG * 4);

  const int NH = NN * HH;            // 320000
  const int nhBlocks = (NH + 255) / 256;

  // 1. pre-swizzle em_W2/em_b2 into WMMA-B order (f16)
  {
    long total = (long)LL * NCHUNK * 4 * 512;
    swizzle_w2<<<(int)((total + 255) / 256), 256, 0, stream>>>(em_W2, em_b2, w2h);
  }
  // 2. input projection (h0 == initial GRU hidden)
  in_proj<<<nhBlocks, 256, 0, stream>>>(x, in_W, in_b, h_cur, hg);
  // 3. zero pooling buffers
  zero_f32<<<(GG * HH + 255) / 256, 256, 0, stream>>>(pool, GG * HH);
  zero_f32<<<1, 256, 0, stream>>>(cnt, GG);

  for (int l = 0; l < LL; ++l) {
    h_to_f16<<<nhBlocks, 256, 0, stream>>>(h_cur, h16);
    edge_mlp<<<(int)(((long)EE * 128 + 255) / 256), 256, 0, stream>>>(
        ea, em_W1 + (size_t)l * 16 * 128, em_b1 + (size_t)l * 128, eact16);
    zero_f32<<<nhBlocks, 256, 0, stream>>>(agg, NH);
    zero_f32<<<1, 256, 0, stream>>>(stats, 128);
    // fused WMMA kernel: 1250 single-wave WGs x 258 iters x 4 WMMAs,
    // B stream double-buffered in LDS via async global->LDS copies
    nnconv_wmma<<<EE / 16, 32, 0, stream>>>(
        h16, eact16, w2h + (size_t)l * NCHUNK * 4 * 512, eidx, agg);
    conv_root<<<(NN + 3) / 4, 256, 0, stream>>>(
        h_cur, agg, root_W + (size_t)l * 64 * 64, conv_b + (size_t)l * 64);
    gru_step<<<NN, 64, 0, stream>>>(agg, hg, gru_Wih, gru_Whh, gru_bih,
                                    gru_bhh, stats);
    bn_relu_res<<<nhBlocks, 256, 0, stream>>>(
        hg, stats, bn_g + (size_t)l * 64, bn_b + (size_t)l * 64, h_cur, h_nxt);
    float* tmp = h_cur; h_cur = h_nxt; h_nxt = tmp;
  }

  pool_accum<<<nhBlocks, 256, 0, stream>>>(h_cur, batch, pool, cnt);
  readout<<<1, 128, 0, stream>>>(pool, cnt, ro_W1, ro_b1, ro_bng, ro_bnb,
                                 ro_W2, ro_b2, out);
}
